// RelationNetworks_78546361909812
// MI455X (gfx1250) — matrix operand order repaired and checked by reading, not compiled, not run on hardware
//
#include <hip/hip_runtime.h>
#include <hip/hip_bf16.h>
#include <stdint.h>

// ---------------------------------------------------------------------------
// RelationNetworks on MI455X (gfx1250, wave32, WMMA).
// Core chain = 3 blocked bf16-WMMA GEMMs with 128x256 workgroup tiles and
// double-buffered LDS staging; h2/h3 intermediates live in the 192MB L2.
// Layer-2 builds h1 on the fly during A-staging; layer-4 fuses the row-sum.
// ---------------------------------------------------------------------------

typedef __attribute__((ext_vector_type(16))) __bf16 v16bf;
typedef __attribute__((ext_vector_type(8)))  float  v8f;

#define NPAIR     2401            // 49*49 pairs per br
#define MTILES    19              // ceil(2401/128)
#define MROWS_BR  2432            // 19*128, padded row stride per br
#define SA_STRIDE 40              // 32 K + 8 pad (bf16) -> conflict-free b128
#define SA_ELEMS  (128 * SA_STRIDE)
#define SB_ELEMS  (256 * SA_STRIDE)

union FragU { v16bf v; uint4 u[2]; };
union Pack8 { __bf16 h[8]; uint4 q; };

// A-fragment (16x32 bf16, MxK) from row-major bf16 (LDS or global).
// ISA: lanes 0-15: M=lane, elems 0-7 -> K=k0..7, elems 8-15 -> K=k0+16..23;
//      lanes 16-31: M=lane-16, K offset +8.
__device__ __forceinline__ v16bf load_a_frag(const __bf16* base, int stride, int k0) {
  int lane = threadIdx.x & 31;
  int m    = lane & 15;
  int kb   = (lane & 16) ? 8 : 0;
  const __bf16* p = base + (size_t)m * stride + (k0 + kb);
  FragU f;
  f.u[0] = *(const uint4*)(p);
  f.u[1] = *(const uint4*)(p + 16);
  return f.v;
}

// B-fragment (32x16 bf16, KxN) from N-major bf16: Bt[n*ldb + k].
// ISA: lanes 0-15: N=lane, elems 0-7 -> K=k0..7, elems 8-15 -> K=k0+16..23;
//      lanes 16-31: N=lane-16, K offset +8.
__device__ __forceinline__ v16bf load_b_frag(const __bf16* Bt, int ldb, int n0, int k0) {
  int lane = threadIdx.x & 31;
  int n    = lane & 15;
  int kb   = (lane & 16) ? 8 : 0;
  const __bf16* p = Bt + (size_t)(n0 + n) * ldb + (k0 + kb);
  FragU f;
  f.u[0] = *(const uint4*)(p);
  f.u[1] = *(const uint4*)(p + 16);
  return f.v;
}

__device__ __forceinline__ v8f wmma_bf16(v16bf a, v16bf b, v8f c) {
  return __builtin_amdgcn_wmma_f32_16x16x32_bf16(false, a, false, b, (short)0, c, false, false);
}

// ---------------------------------------------------------------------------
// Tiled transpose + f32 -> bf16 convert:  Wt[n*Kpad + k] = (k<K) ? W[k*ldw+n] : 0
// ---------------------------------------------------------------------------
__global__ void transpose_cvt(const float* __restrict__ W, __bf16* __restrict__ Wt,
                              int K, int N, int ldw, int Kpad) {
  __shared__ float tile[32][33];
  int kb = blockIdx.x * 32, nb = blockIdx.y * 32;
  for (int i = 0; i < 32; i += 8) {
    int k = kb + threadIdx.y + i, n = nb + threadIdx.x;
    tile[threadIdx.y + i][threadIdx.x] = (k < K && n < N) ? W[(size_t)k * ldw + n] : 0.f;
  }
  __syncthreads();
  for (int i = 0; i < 32; i += 8) {
    int n = nb + threadIdx.y + i, k = kb + threadIdx.x;
    if (n < N && k < Kpad) Wt[(size_t)n * Kpad + k] = (__bf16)tile[threadIdx.x][threadIdx.y + i];
  }
}

// ---------------------------------------------------------------------------
// Build padded bf16 feat matrix: 112 rows (98 valid = 2 batches x 49 positions),
// 544 cols (514 valid = 512 channels + 2 coords).
// ---------------------------------------------------------------------------
__global__ void build_featpad(const float* __restrict__ conv, __bf16* __restrict__ fp) {
  int idx = blockIdx.x * 256 + threadIdx.x;
  if (idx >= 112 * 544) return;
  int r = idx / 544, k = idx - r * 544;
  float v = 0.f;
  if (r < 98 && k < 514) {
    int b = r / 49, p = r - b * 49;
    if (k < 512) {
      v = conv[(size_t)b * 25088 + (size_t)k * 49 + p];
    } else {
      float step = 8.f / 6.f;
      v = (k == 512) ? (-4.f + step * (p % 7)) : (-4.f + step * (p / 7));
    }
  }
  fp[idx] = (__bf16)v;
}

// rv[b*6+r, e] = verb_emb[verbs[b], e] * role_emb[roles[b,r], e]   (12 x 512, f32)
__global__ void build_rv(const float* __restrict__ verb_emb, const float* __restrict__ role_emb,
                         const int* __restrict__ verbs, const int* __restrict__ roles,
                         float* __restrict__ rv) {
  int idx = blockIdx.x * 256 + threadIdx.x;
  if (idx >= 12 * 512) return;
  int br = idx >> 9, e = idx & 511;
  int b = br / 6, r = br - b * 6;
  rv[idx] = verb_emb[(size_t)verbs[b] * 512 + e] * role_emb[(size_t)roles[b * 6 + r] * 512 + e];
}

__global__ void zero_f32(float* p, int n) {
  int i = blockIdx.x * 256 + threadIdx.x;
  if (i < n) p[i] = 0.f;
}

// ---------------------------------------------------------------------------
// Small WMMA GEMM (feat projections): C_f32[M x N] = A_bf16 @ Bt_bf16^T
// ---------------------------------------------------------------------------
__global__ __launch_bounds__(256) void wmma_gemm_f32(
    const __bf16* __restrict__ A, int lda,
    const __bf16* __restrict__ Bt, int ldb,
    float* __restrict__ C, int ldc, int K, int N) {
  int wave = threadIdx.x >> 5;
  int n0 = blockIdx.y * 128 + wave * 16;
  if (n0 >= N) return;   // wave-uniform
  const __bf16* Ab = A + (size_t)blockIdx.x * 16 * lda;
  v8f acc = {0.f, 0.f, 0.f, 0.f, 0.f, 0.f, 0.f, 0.f};
  for (int k0 = 0; k0 < K; k0 += 32) {
    v16bf a = load_a_frag(Ab, lda, k0);
    v16bf b = load_b_frag(Bt, ldb, n0, k0);
    acc = wmma_bf16(a, b, acc);
  }
  int lane = threadIdx.x & 31;
  int crow = (lane & 16) ? 8 : 0;
  int ccol = lane & 15;
#pragma unroll
  for (int v = 0; v < 8; ++v)
    C[(size_t)(blockIdx.x * 16 + crow + v) * ldc + n0 + ccol] = acc[v];
}

// ---------------------------------------------------------------------------
// Skinny fp32 GEMM (memory-bound, M<=16): out[m,n] = relu?(X[m,:]@W[:,n]+bias)
// ---------------------------------------------------------------------------
template <int MR>
__global__ __launch_bounds__(256) void skinny_gemm(
    const float* __restrict__ X, const float* __restrict__ W,
    const float* __restrict__ bias, float* __restrict__ out,
    int K, int N, int ldw, int ldc, int do_relu) {
  __shared__ float sx[MR * 256];
  int n = blockIdx.x * 256 + threadIdx.x;
  float acc[MR];
#pragma unroll
  for (int m = 0; m < MR; ++m) acc[m] = 0.f;

  for (int k0 = 0; k0 < K; k0 += 256) {
    int kt = (K - k0 < 256) ? (K - k0) : 256;
    __syncthreads();
    for (int i = threadIdx.x; i < MR * kt; i += 256) {
      int m = i / kt, kk = i - m * kt;
      sx[m * 256 + kk] = X[(size_t)m * K + k0 + kk];
    }
    __syncthreads();
    if (n < N) {
      for (int kk = 0; kk < kt; ++kk) {
        float wv = W[(size_t)(k0 + kk) * ldw + n];
#pragma unroll
        for (int m = 0; m < MR; ++m) acc[m] += sx[m * 256 + kk] * wv;
      }
    }
  }
  if (n < N) {
    float bz = bias ? bias[n] : 0.f;
#pragma unroll
    for (int m = 0; m < MR; ++m) {
      float r = acc[m] + bz;
      if (do_relu) r = fmaxf(r, 0.f);
      out[(size_t)m * ldc + n] = r;
    }
  }
}

// ---------------------------------------------------------------------------
// Blocked WMMA GEMM, 128(M) x 256(N) workgroup tile, 512 threads = 16 waves
// in a 4x4 grid; each wave computes a 32x64 C-tile (2x4 v8f accumulators).
// K staged in 32-deep double-buffered LDS panels (stride-40 rows, conflict-
// free b128). grid = (MTILES, N/256, 12 br).
//   AMODE 0: A from global bf16 (lda, per-br stride)
//   AMODE 1: A = h1 built on the fly: relu(Aout[b,z]+Bm[b,y]+Q[br]+gb1)
//   OMODE 0: store bf16 relu(acc+bias) to Hout (per-br stride)
//   OMODE 1: fold rows: gsum[br,n] += sum_rows relu(acc+bias)   (h4 virtual)
// ---------------------------------------------------------------------------
template <int AMODE, int OMODE>
__global__ __launch_bounds__(512) void gemm_tile(
    const __bf16* __restrict__ Abase, int lda, size_t a_br_stride,
    const float* __restrict__ Aout, const float* __restrict__ Bmout,
    const float* __restrict__ Qout, const float* __restrict__ gb1,
    const __bf16* __restrict__ Bt, int ldb,
    const float* __restrict__ bias,
    __bf16* __restrict__ Hout, int ldc, size_t h_br_stride,
    float* __restrict__ gsum, int K, int Mvalid) {
  __shared__ __align__(16) __bf16 sA[2 * SA_ELEMS];   // 2 x 128x40 bf16 = 20 KB
  __shared__ __align__(16) __bf16 sB[2 * SB_ELEMS];   // 2 x 256x40 bf16 = 40 KB

  int tid = threadIdx.x;
  int br  = blockIdx.z;
  int b   = br / 6;
  int r0  = blockIdx.x * 128;
  int nb0 = blockIdx.y * 256;

  const __bf16* Ab = Abase + (size_t)br * a_br_stride;

  // staging roles
  int arow = tid >> 2, aq = tid & 3;      // A: 128 rows x 4 chunks of 8 elems
  int bcol = tid >> 1, bh = tid & 1;      // B: 256 cols x 2 chunks of 16 elems

  // wave tiling
  int wave = tid >> 5;
  int wy = wave >> 2;                     // 0..3 -> rows wy*32
  int wx = wave & 3;                      // 0..3 -> cols wx*64
  int lane = tid & 31;
  int crow = (lane & 16) ? 8 : 0;
  int ccol = lane & 15;

  v8f acc[2][4];
#pragma unroll
  for (int fm = 0; fm < 2; ++fm)
#pragma unroll
    for (int fn = 0; fn < 4; ++fn)
      acc[fm][fn] = (v8f){0.f, 0.f, 0.f, 0.f, 0.f, 0.f, 0.f, 0.f};

  const uint4 zero4 = {0u, 0u, 0u, 0u};
  uint4 ra, rb0, rb1;

  // ---- register prefetch of K-panel 0 ----
  {
    int rA = r0 + arow;
    if (AMODE == 0)
      ra = (rA < Mvalid) ? *(const uint4*)(Ab + (size_t)rA * lda + aq * 8) : zero4;
    const __bf16* pB = Bt + (size_t)(nb0 + bcol) * ldb + bh * 16;
    rb0 = ((const uint4*)pB)[0];
    rb1 = ((const uint4*)pB)[1];
  }

  int nT = K >> 5;
  for (int t = 0; t < nT; ++t) {
    int kt = t << 5;
    __bf16* sAb = sA + (t & 1) * SA_ELEMS;
    __bf16* sBb = sB + (t & 1) * SB_ELEMS;

    // ---- commit staged panel to LDS ----
    if (AMODE == 1) {
      int rA = r0 + arow;
      Pack8 pk;
      if (rA < Mvalid) {
        int y = rA / 49, z = rA - y * 49;
        const float* pa_ = Aout + (size_t)(b * 49 + z) * 1024 + kt + aq * 8;
        const float* pb_ = Bmout + (size_t)(b * 49 + y) * 1024 + kt + aq * 8;
        const float* pq_ = Qout + (size_t)br * 1024 + kt + aq * 8;
        const float* pg_ = gb1 + kt + aq * 8;
#pragma unroll
        for (int e = 0; e < 8; ++e)
          pk.h[e] = (__bf16)fmaxf(pa_[e] + pb_[e] + pq_[e] + pg_[e], 0.f);
      } else {
        pk.q = zero4;
      }
      *(uint4*)(sAb + arow * SA_STRIDE + aq * 8) = pk.q;
    } else {
      *(uint4*)(sAb + arow * SA_STRIDE + aq * 8) = ra;
    }
    *(uint4*)(sBb + bcol * SA_STRIDE + bh * 16)     = rb0;
    *(uint4*)(sBb + bcol * SA_STRIDE + bh * 16 + 8) = rb1;

    // ---- register prefetch of next K-panel (overlaps with compute) ----
    if (t + 1 < nT) {
      int kn = kt + 32;
      if (AMODE == 0) {
        int rA = r0 + arow;
        ra = (rA < Mvalid) ? *(const uint4*)(Ab + (size_t)rA * lda + kn + aq * 8) : zero4;
      }
      const __bf16* pB = Bt + (size_t)(nb0 + bcol) * ldb + kn + bh * 16;
      rb0 = ((const uint4*)pB)[0];
      rb1 = ((const uint4*)pB)[1];
    }

    __syncthreads();

    // ---- compute: 2 A-frags x 4 B-frags -> 8 WMMAs ----
    v16bf a0 = load_a_frag(sAb + (wy * 32 + 0) * SA_STRIDE, SA_STRIDE, 0);
    v16bf a1 = load_a_frag(sAb + (wy * 32 + 16) * SA_STRIDE, SA_STRIDE, 0);
#pragma unroll
    for (int fn = 0; fn < 4; ++fn) {
      v16bf bb = load_b_frag(sBb, SA_STRIDE, wx * 64 + fn * 16, 0);
      acc[0][fn] = wmma_bf16(a0, bb, acc[0][fn]);
      acc[1][fn] = wmma_bf16(a1, bb, acc[1][fn]);
    }
  }

  // ---- epilogue ----
  if (OMODE == 0) {
    __bf16* Ho = Hout + (size_t)br * h_br_stride;
#pragma unroll
    for (int fn = 0; fn < 4; ++fn) {
      int col = nb0 + wx * 64 + fn * 16 + ccol;
      float bz = bias[col];
#pragma unroll
      for (int fm = 0; fm < 2; ++fm) {
#pragma unroll
        for (int v = 0; v < 8; ++v) {
          int row = r0 + wy * 32 + fm * 16 + crow + v;
          if (row < Mvalid)
            Ho[(size_t)row * ldc + col] = (__bf16)fmaxf(acc[fm][fn][v] + bz, 0.f);
        }
      }
    }
  } else {
#pragma unroll
    for (int fn = 0; fn < 4; ++fn) {
      int col = nb0 + wx * 64 + fn * 16 + ccol;
      float bz = bias[col];
      float s = 0.f;
#pragma unroll
      for (int fm = 0; fm < 2; ++fm) {
#pragma unroll
        for (int v = 0; v < 8; ++v) {
          int row = r0 + wy * 32 + fm * 16 + crow + v;
          float val = fmaxf(acc[fm][fn][v] + bz, 0.f);
          s += (row < Mvalid) ? val : 0.f;
        }
      }
      s += __shfl_xor(s, 16);           // fold row-halves sharing the column
      if (lane < 16)
        atomicAdd(&gsum[(size_t)br * 4096 + nb0 + wx * 64 + fn * 16 + lane], s);
    }
  }
}

// ---------------------------------------------------------------------------
// Host orchestration
// ---------------------------------------------------------------------------
extern "C" void kernel_launch(void* const* d_in, const int* in_sizes, int n_in,
                              void* d_out, int out_size, void* d_ws, size_t ws_size,
                              hipStream_t stream) {
  const float* conv     = (const float*)d_in[0];
  const int*   verbs    = (const int*)  d_in[1];
  const int*   roles    = (const int*)  d_in[2];
  const float* verb_emb = (const float*)d_in[3];
  const float* role_emb = (const float*)d_in[4];
  const float* vW1 = (const float*)d_in[5];  const float* vb1 = (const float*)d_in[6];
  const float* vW2 = (const float*)d_in[7];  const float* vb2 = (const float*)d_in[8];
  const float* vW3 = (const float*)d_in[9];  const float* vb3 = (const float*)d_in[10];
  const float* gW1 = (const float*)d_in[11]; const float* gb1 = (const float*)d_in[12];
  const float* gW2 = (const float*)d_in[13]; const float* gb2 = (const float*)d_in[14];
  const float* gW3 = (const float*)d_in[15]; const float* gb3 = (const float*)d_in[16];
  const float* gW4 = (const float*)d_in[17]; const float* gb4 = (const float*)d_in[18];
  const float* fW1 = (const float*)d_in[19]; const float* fb1 = (const float*)d_in[20];
  const float* fW2 = (const float*)d_in[21]; const float* fb2 = (const float*)d_in[22];
  const float* fW3 = (const float*)d_in[23]; const float* fb3 = (const float*)d_in[24];
  float* out = (float*)d_out;   // [0,1000): verb_pred ; [1000,25000): role_predict

  // workspace layout (256B aligned); h2/h3 are L2-resident staging tensors
  size_t off = 0;
  auto alloc = [&](size_t bytes) { size_t o = off; off += (bytes + 255) & ~(size_t)255; return o; };
  char* ws = (char*)d_ws;
  __bf16* gW2t    = (__bf16*)(ws + alloc((size_t)1024 * 1024 * 2));  // [N=1024][K=1024]
  __bf16* gW3t    = (__bf16*)(ws + alloc((size_t)2048 * 1024 * 2));  // [N=2048][K=1024]
  __bf16* gW4t    = (__bf16*)(ws + alloc((size_t)4096 * 2048 * 2));  // [N=4096][K=2048]
  __bf16* Wat     = (__bf16*)(ws + alloc((size_t)1024 * 544 * 2));   // [N=1024][Kpad=544]
  __bf16* Wbt     = (__bf16*)(ws + alloc((size_t)1024 * 544 * 2));
  __bf16* featpad = (__bf16*)(ws + alloc((size_t)112 * 544 * 2));
  float*  Aout    = (float*) (ws + alloc((size_t)112 * 1024 * 4));
  float*  Bmout   = (float*) (ws + alloc((size_t)112 * 1024 * 4));
  float*  Qout    = (float*) (ws + alloc((size_t)12 * 1024 * 4));
  float*  rv      = (float*) (ws + alloc((size_t)12 * 512 * 4));
  float*  gsum    = (float*) (ws + alloc((size_t)12 * 4096 * 4));
  float*  v1      = (float*) (ws + alloc((size_t)2 * 1024 * 4));
  float*  v2      = (float*) (ws + alloc((size_t)2 * 1024 * 4));
  float*  f1      = (float*) (ws + alloc((size_t)12 * 4096 * 4));
  float*  f2      = (float*) (ws + alloc((size_t)12 * 2048 * 4));
  __bf16* h2buf   = (__bf16*)(ws + alloc((size_t)12 * MROWS_BR * 1024 * 2)); // ~60 MB
  __bf16* h3buf   = (__bf16*)(ws + alloc((size_t)12 * MROWS_BR * 2048 * 2)); // ~120 MB
  (void)ws_size; (void)n_in; (void)in_sizes; (void)out_size;

  dim3 tb(32, 8);
  // weight convert+transpose to bf16 [N][Kpad]
  transpose_cvt<<<dim3(32, 32),  tb, 0, stream>>>(gW2, gW2t, 1024, 1024, 1024, 1024);
  transpose_cvt<<<dim3(32, 64),  tb, 0, stream>>>(gW3, gW3t, 1024, 2048, 2048, 1024);
  transpose_cvt<<<dim3(64, 128), tb, 0, stream>>>(gW4, gW4t, 2048, 4096, 4096, 2048);
  transpose_cvt<<<dim3(17, 32),  tb, 0, stream>>>(gW1,              Wat, 514, 1024, 1024, 544);
  transpose_cvt<<<dim3(17, 32),  tb, 0, stream>>>(gW1 + 514 * 1024, Wbt, 514, 1024, 1024, 544);

  build_featpad<<<(112 * 544 + 255) / 256, 256, 0, stream>>>(conv, featpad);
  build_rv<<<(12 * 512 + 255) / 256, 256, 0, stream>>>(verb_emb, role_emb, verbs, roles, rv);
  zero_f32<<<(12 * 4096 + 255) / 256, 256, 0, stream>>>(gsum, 12 * 4096);

  // A = feat @ Wa, Bm = feat @ Wb  (M=112 pad, K=544 pad, N=1024) via WMMA
  wmma_gemm_f32<<<dim3(7, 8), 256, 0, stream>>>(featpad, 544, Wat, 544, Aout, 1024, 544, 1024);
  wmma_gemm_f32<<<dim3(7, 8), 256, 0, stream>>>(featpad, 544, Wbt, 544, Bmout, 1024, 544, 1024);

  // Q = rv @ Wc (Wc = gW1 rows [1028,1540)), M=12, K=512, N=1024
  skinny_gemm<12><<<4, 256, 0, stream>>>(rv, gW1 + (size_t)2 * 514 * 1024, nullptr, Qout,
                                         512, 1024, 1024, 1024, 0);

  // verb MLP (M=2, weight-read bound)
  skinny_gemm<2><<<4, 256, 0, stream>>>(conv, vW1, vb1, v1, 25088, 1024, 1024, 1024, 1);
  skinny_gemm<2><<<4, 256, 0, stream>>>(v1,   vW2, vb2, v2, 1024, 1024, 1024, 1024, 1);
  skinny_gemm<2><<<2, 256, 0, stream>>>(v2,   vW3, vb3, out, 1024, 500, 500, 500, 0);

  // ---- the 665 GFLOP chain: three blocked WMMA GEMMs over 28812 rows ----
  // layer 2: h2 = relu(h1 @ gW2 + gb2); h1 built in A-staging
  gemm_tile<1, 0><<<dim3(MTILES, 4, 12), 512, 0, stream>>>(
      nullptr, 0, 0, Aout, Bmout, Qout, gb1,
      gW2t, 1024, gb2, h2buf, 1024, (size_t)MROWS_BR * 1024, nullptr, 1024, NPAIR);
  // layer 3: h3 = relu(h2 @ gW3 + gb3)
  gemm_tile<0, 0><<<dim3(MTILES, 8, 12), 512, 0, stream>>>(
      h2buf, 1024, (size_t)MROWS_BR * 1024, nullptr, nullptr, nullptr, nullptr,
      gW3t, 1024, gb3, h3buf, 2048, (size_t)MROWS_BR * 2048, nullptr, 1024, NPAIR);
  // layer 4: gsum[br] += sum_rows relu(h3 @ gW4 + gb4)
  gemm_tile<0, 1><<<dim3(MTILES, 16, 12), 512, 0, stream>>>(
      h3buf, 2048, (size_t)MROWS_BR * 2048, nullptr, nullptr, nullptr, nullptr,
      gW4t, 2048, gb4, nullptr, 0, 0, gsum, 2048, NPAIR);

  // f-MLP (M=12, weight-read bound)
  skinny_gemm<12><<<16, 256, 0, stream>>>(gsum, fW1, fb1, f1, 4096, 4096, 4096, 4096, 1);
  skinny_gemm<12><<<8,  256, 0, stream>>>(f1,   fW2, fb2, f2, 4096, 2048, 2048, 2048, 1);
  skinny_gemm<12><<<8,  256, 0, stream>>>(f2,   fW3, fb3, out + 1000, 2048, 2000, 2000, 2000, 0);
}
